// downstream2_26285199852192
// MI455X (gfx1250) — matrix operand order repaired and checked by reading, not compiled, not run on hardware
//
#include <hip/hip_runtime.h>
#include <hip/hip_bf16.h>
#include <math.h>

#define N_NODES  20000
#define N_EDGES  320000
#define F_EDGE   32
#define D_IN     16
#define D_NN     16
#define D_GAT    64
#define HID      128
#define N_GRAPHS 64
#define M_ENTRIES (N_EDGES + N_NODES)   // edges + self loops

typedef __attribute__((ext_vector_type(16))) _Float16 v16h;
typedef __attribute__((ext_vector_type(8)))  float    v8f;
typedef __attribute__((ext_vector_type(2)))  _Float16 h2;
typedef __attribute__((ext_vector_type(2)))  __fp16   fh2;

union V16 { v16h v; h2 h[8]; };

// pack two f32 -> f16x2 (v_cvt_pk_rtz_f16_f32, round toward zero), bit-cast to _Float16 vector type
static __device__ __forceinline__ h2 pk2(float a, float b) {
    fh2 t = __builtin_amdgcn_cvt_pkrtz(a, b);
    return __builtin_bit_cast(h2, t);
}

// ---------------------------------------------------------------------------
// Kernel 0: initialize accumulators (harness does not re-zero between replays)
// ---------------------------------------------------------------------------
__global__ void init_kernel(float* __restrict__ agg, float* __restrict__ denom,
                            float* __restrict__ mbuf, float* __restrict__ gat,
                            float* __restrict__ pooled, float* __restrict__ cnt) {
    int i = blockIdx.x * blockDim.x + threadIdx.x;
    int stride = gridDim.x * blockDim.x;
    for (int j = i; j < N_NODES * D_GAT; j += stride) gat[j] = 0.f;
    for (int j = i; j < N_NODES * D_NN;  j += stride) agg[j] = 0.f;
    for (int j = i; j < N_NODES; j += stride) {
        denom[j] = 0.f;
        ((unsigned int*)mbuf)[j] = 0xFF800000u;   // -inf
    }
    for (int j = i; j < N_GRAPHS * D_GAT; j += stride) pooled[j] = 0.f;
    for (int j = i; j < N_GRAPHS; j += stride) cnt[j] = 0.f;
}

// ---------------------------------------------------------------------------
// Kernel 1: fused NNConv message = (edge_attr @ We + be) contracted with x[src],
// computed as Z @ W2 via v_wmma_f32_16x16x32_f16, scattered to agg[dst] with
// global f32 atomics.  Z[e][k*16+i] = ea[e][k] * x[src_e][i];  W2 == We flat.
// One wave handles one 16-edge tile per iteration; B (We as f16) is resident
// in 17*8 VGPRs.  EXEC is all-ones for every WMMA (uniform control flow).
// ---------------------------------------------------------------------------
__global__ void __launch_bounds__(256)
nnconv_wmma_kernel(const float* __restrict__ x, const int* __restrict__ ei,
                   const float* __restrict__ ea, const float* __restrict__ We,
                   const float* __restrict__ be, float* __restrict__ agg) {
    const int lane = threadIdx.x & 31;
    const int wave = (blockIdx.x * blockDim.x + threadIdx.x) >> 5;
    const int nwaves = (gridDim.x * blockDim.x) >> 5;
    const int col = lane & 15;          // A-row edge within tile; also C column (o)
    const int hi  = lane >> 4;          // lane half: selects K/i sub-range

    // ---- B operand: We viewed as [512][16] f32 -> f16 in WMMA B layout ----
    // chunk t, VGPR v, lane: B[K=8*hi+2v+(v>=4?8:0)+{0,1}][n=col] with global row
    // g = 32*t + K  ->  element We_flat[g*16 + col].
    V16 B[17];
#pragma unroll
    for (int t = 0; t < 16; ++t) {
#pragma unroll
        for (int v = 0; v < 8; ++v) {
            int g0 = 32 * t + 8 * hi + 2 * v + 8 * (v >> 2);
            B[t].h[v] = pk2(We[g0 * 16 + col], We[(g0 + 1) * 16 + col]);
        }
    }
    // bias chunk: elements s<8 hold be[(8*hi+s)*16+n], elements s>=8 zero
#pragma unroll
    for (int v = 0; v < 8; ++v) {
        float b0 = (v < 4) ? be[(8 * hi + 2 * v) * 16 + col]     : 0.f;
        float b1 = (v < 4) ? be[(8 * hi + 2 * v + 1) * 16 + col] : 0.f;
        B[16].h[v] = pk2(b0, b1);
    }

    const int ntiles = N_EDGES / 16;
    for (int tile = wave; tile < ntiles; tile += nwaves) {
        const int base = tile * 16;
        const int e = base + col;                 // this lane's A-matrix edge

        // prefetch next tile's edge_attr rows into cache
        if (tile + 1 < ntiles)
            __builtin_prefetch(&ea[(size_t)(e + 16) * F_EDGE], 0, 1);

        // gather x[src_e][i0 .. i0+7]  (i0 = 0 for lanes 0-15, 8 for 16-31)
        const int src = ei[e];                    // edge_index[0][e]
        const float4 xa = *(const float4*)&x[src * D_IN + hi * 8 + 0];
        const float4 xb = *(const float4*)&x[src * D_IN + hi * 8 + 4];
        h2 xp[4];
        xp[0] = pk2(xa.x, xa.y);
        xp[1] = pk2(xa.z, xa.w);
        xp[2] = pk2(xb.x, xb.y);
        xp[3] = pk2(xb.z, xb.w);

        // edge_attr row (32 f32) -> 16 packed f16 pairs
        h2 eap[16];
#pragma unroll
        for (int q = 0; q < 8; ++q) {
            float4 t4 = *(const float4*)&ea[(size_t)e * F_EDGE + q * 4];
            eap[2 * q + 0] = pk2(t4.x, t4.y);
            eap[2 * q + 1] = pk2(t4.z, t4.w);
        }

        v8f c = {};
#pragma unroll
        for (int t = 0; t < 16; ++t) {
            _Float16 alo = eap[t].x;   // ea[e][2t]
            _Float16 ahi = eap[t].y;   // ea[e][2t+1]
            V16 A;
#pragma unroll
            for (int v = 0; v < 4; ++v) {
                A.h[v]     = alo * xp[v];   // K pairs with k=2t
                A.h[v + 4] = ahi * xp[v];   // K pairs with k=2t+1
            }
            c = __builtin_amdgcn_wmma_f32_16x16x32_f16(
                    false, A.v, false, B[t].v, (short)0, c, false, false);
        }
        {   // bias chunk: A = {x pairs, zeros}
            h2 z2 = pk2(0.f, 0.f);
            V16 A;
#pragma unroll
            for (int v = 0; v < 4; ++v) { A.h[v] = xp[v]; A.h[v + 4] = z2; }
            c = __builtin_amdgcn_wmma_f32_16x16x32_f16(
                    false, A.v, false, B[16].v, (short)0, c, false, false);
        }

        // scatter: C VGPR r holds edge m = hi*8 + r, column o = col
#pragma unroll
        for (int r = 0; r < 8; ++r) {
            int m = hi * 8 + r;
            int d = ei[N_EDGES + base + m];       // edge_index[1][...]
            atomicAdd(&agg[d * D_NN + col], c[r]);
        }
    }
}

// ---------------------------------------------------------------------------
// Kernel 2: h = relu(agg + x@Wroot + bconv)   (in-place into agg)
// ---------------------------------------------------------------------------
__global__ void root_relu_kernel(const float* __restrict__ x,
                                 const float* __restrict__ Wroot,
                                 const float* __restrict__ bconv,
                                 float* __restrict__ agg) {
    int id = blockIdx.x * blockDim.x + threadIdx.x;
    if (id >= N_NODES * D_NN) return;
    int n = id >> 4, o = id & 15;
    float s = bconv[o];
#pragma unroll
    for (int i = 0; i < D_IN; ++i) s += x[n * D_IN + i] * Wroot[i * D_NN + o];
    float v = agg[id] + s;
    agg[id] = v > 0.f ? v : 0.f;
}

// ---------------------------------------------------------------------------
// Kernel 3: hp = h @ Wgat
// ---------------------------------------------------------------------------
__global__ void gatproj_kernel(const float* __restrict__ h,
                               const float* __restrict__ Wgat,
                               float* __restrict__ hp) {
    int id = blockIdx.x * blockDim.x + threadIdx.x;
    if (id >= N_NODES * D_GAT) return;
    int n = id >> 6, o = id & 63;
    float s = 0.f;
#pragma unroll
    for (int i = 0; i < D_NN; ++i) s += h[n * D_NN + i] * Wgat[i * D_GAT + o];
    hp[id] = s;
}

// ---------------------------------------------------------------------------
// Kernel 3b: per-node attention score halves
// ---------------------------------------------------------------------------
__global__ void score_kernel(const float* __restrict__ hp,
                             const float* __restrict__ a_src,
                             const float* __restrict__ a_dst,
                             float* __restrict__ sc_s, float* __restrict__ sc_d) {
    int n = blockIdx.x * blockDim.x + threadIdx.x;
    if (n >= N_NODES) return;
    float ss = 0.f, sd = 0.f;
#pragma unroll 8
    for (int o = 0; o < D_GAT; ++o) {
        float v = hp[n * D_GAT + o];
        ss += v * a_src[o];
        sd += v * a_dst[o];
    }
    sc_s[n] = ss; sc_d[n] = sd;
}

// ---------------------------------------------------------------------------
// Kernel 4a: e = leaky_relu(sc_s[s]+sc_d[d]); segment_max via float-atomic trick
// ---------------------------------------------------------------------------
__global__ void attn_max_kernel(const int* __restrict__ ei,
                                const float* __restrict__ sc_s,
                                const float* __restrict__ sc_d,
                                float* __restrict__ esc, float* __restrict__ mbuf) {
    int t = blockIdx.x * blockDim.x + threadIdx.x;
    if (t >= M_ENTRIES) return;
    int s, d;
    if (t < N_EDGES) { s = ei[t]; d = ei[N_EDGES + t]; }
    else             { s = d = t - N_EDGES; }
    float e = sc_s[s] + sc_d[d];
    e = e > 0.f ? e : 0.2f * e;
    esc[t] = e;
    if (e >= 0.f) atomicMax((int*)&mbuf[d], __float_as_int(e));
    else          atomicMin((unsigned int*)&mbuf[d], __float_as_uint(e));
}

// ---------------------------------------------------------------------------
// Kernel 4b: ex = exp(e - m[d]); denom[d] += ex   (esc overwritten in place)
// ---------------------------------------------------------------------------
__global__ void attn_exp_kernel(const int* __restrict__ ei,
                                const float* __restrict__ mbuf,
                                float* __restrict__ esc, float* __restrict__ denom) {
    int t = blockIdx.x * blockDim.x + threadIdx.x;
    if (t >= M_ENTRIES) return;
    int d = (t < N_EDGES) ? ei[N_EDGES + t] : (t - N_EDGES);
    float ex = __expf(esc[t] - mbuf[d]);
    esc[t] = ex;
    atomicAdd(&denom[d], ex);
}

// ---------------------------------------------------------------------------
// Kernel 4c: gat[d] += alpha * hp[s]   (thread = entry x 16-wide column slab)
// ---------------------------------------------------------------------------
__global__ void attn_scatter_kernel(const int* __restrict__ ei,
                                    const float* __restrict__ esc,
                                    const float* __restrict__ denom,
                                    const float* __restrict__ hp,
                                    float* __restrict__ gat) {
    int id = blockIdx.x * blockDim.x + threadIdx.x;
    int t = id >> 2;
    if (t >= M_ENTRIES) return;
    int o0 = (id & 3) * 16;
    int s, d;
    if (t < N_EDGES) { s = ei[t]; d = ei[N_EDGES + t]; }
    else             { s = d = t - N_EDGES; }
    float alpha = esc[t] / denom[d];
#pragma unroll
    for (int o = o0; o < o0 + 16; ++o)
        atomicAdd(&gat[d * D_GAT + o], alpha * hp[s * D_GAT + o]);
}

// ---------------------------------------------------------------------------
// Kernel 5: relu(gat + bgat) then global mean-pool numerators + counts
// ---------------------------------------------------------------------------
__global__ void pool_kernel(const float* __restrict__ gat,
                            const float* __restrict__ bgat,
                            const int* __restrict__ batch,
                            float* __restrict__ pooled, float* __restrict__ cnt) {
    int id = blockIdx.x * blockDim.x + threadIdx.x;
    if (id >= N_NODES * D_GAT) return;
    int n = id >> 6, o = id & 63;
    float g = gat[id] + bgat[o];
    g = g > 0.f ? g : 0.f;
    int b = batch[n];
    atomicAdd(&pooled[b * D_GAT + o], g);
    if (o == 0) atomicAdd(&cnt[b], 1.0f);
}

// ---------------------------------------------------------------------------
// Kernel 6: MLP head -> logits (single 128-thread block, G=64 graphs)
// ---------------------------------------------------------------------------
__global__ void head_kernel(const float* __restrict__ pooled,
                            const float* __restrict__ cnt,
                            const float* __restrict__ Wfc1,
                            const float* __restrict__ bfc1,
                            const float* __restrict__ Wfc2,
                            const float* __restrict__ bfc2,
                            float* __restrict__ out) {
    __shared__ float pa[D_GAT];
    __shared__ float red[HID];
    int tid = threadIdx.x;
    for (int g = 0; g < N_GRAPHS; ++g) {
        if (tid < D_GAT) pa[tid] = pooled[g * D_GAT + tid] / fmaxf(cnt[g], 1.0f);
        __syncthreads();
        float z = bfc1[tid];
#pragma unroll 8
        for (int o = 0; o < D_GAT; ++o) z += pa[o] * Wfc1[o * HID + tid];
        z = z > 0.f ? z : 0.f;
        red[tid] = z * Wfc2[tid];
        __syncthreads();
        for (int s2 = HID / 2; s2 > 0; s2 >>= 1) {
            if (tid < s2) red[tid] += red[tid + s2];
            __syncthreads();
        }
        if (tid == 0) out[g] = red[0] + bfc2[0];
        __syncthreads();
    }
}

// ---------------------------------------------------------------------------
extern "C" void kernel_launch(void* const* d_in, const int* in_sizes, int n_in,
                              void* d_out, int out_size, void* d_ws, size_t ws_size,
                              hipStream_t stream) {
    const float* x     = (const float*)d_in[0];
    const int*   ei    = (const int*)  d_in[1];   // [2, E]
    const float* ea    = (const float*)d_in[2];
    const int*   batch = (const int*)  d_in[3];
    const float* We    = (const float*)d_in[4];
    const float* be    = (const float*)d_in[5];
    const float* Wroot = (const float*)d_in[6];
    const float* bconv = (const float*)d_in[7];
    const float* Wgat  = (const float*)d_in[8];
    const float* a_src = (const float*)d_in[9];
    const float* a_dst = (const float*)d_in[10];
    const float* bgat  = (const float*)d_in[11];
    const float* Wfc1  = (const float*)d_in[12];
    const float* bfc1  = (const float*)d_in[13];
    const float* Wfc2  = (const float*)d_in[14];
    const float* bfc2  = (const float*)d_in[15];
    float* out = (float*)d_out;

    // workspace layout (floats)
    float* ws     = (float*)d_ws;
    float* agg    = ws;                         // N*16   (becomes h)
    float* hp     = agg    + N_NODES * D_NN;    // N*64
    float* sc_s   = hp     + N_NODES * D_GAT;   // N
    float* sc_d   = sc_s   + N_NODES;           // N
    float* mbuf   = sc_d   + N_NODES;           // N
    float* denom  = mbuf   + N_NODES;           // N
    float* esc    = denom  + N_NODES;           // E+N   (scores, then exp)
    float* gat    = esc    + M_ENTRIES;         // N*64
    float* pooled = gat    + N_NODES * D_GAT;   // G*64
    float* cnt    = pooled + N_GRAPHS * D_GAT;  // G

    init_kernel<<<2048, 256, 0, stream>>>(agg, denom, mbuf, gat, pooled, cnt);

    nnconv_wmma_kernel<<<200, 256, 0, stream>>>(x, ei, ea, We, be, agg);

    root_relu_kernel<<<(N_NODES * D_NN + 255) / 256, 256, 0, stream>>>(x, Wroot, bconv, agg);

    gatproj_kernel<<<(N_NODES * D_GAT + 255) / 256, 256, 0, stream>>>(agg, Wgat, hp);

    score_kernel<<<(N_NODES + 255) / 256, 256, 0, stream>>>(hp, a_src, a_dst, sc_s, sc_d);

    attn_max_kernel<<<(M_ENTRIES + 255) / 256, 256, 0, stream>>>(ei, sc_s, sc_d, esc, mbuf);

    attn_exp_kernel<<<(M_ENTRIES + 255) / 256, 256, 0, stream>>>(ei, mbuf, esc, denom);

    attn_scatter_kernel<<<(M_ENTRIES * 4 + 255) / 256, 256, 0, stream>>>(ei, esc, denom, hp, gat);

    pool_kernel<<<(N_NODES * D_GAT + 255) / 256, 256, 0, stream>>>(gat, bgat, batch, pooled, cnt);

    head_kernel<<<1, HID, 0, stream>>>(pooled, cnt, Wfc1, bfc1, Wfc2, bfc2, out);
}
